// Bi_LSTM_80805514707581
// MI455X (gfx1250) — matrix operand order repaired and checked by reading, not compiled, not run on hardware
//
#include <hip/hip_runtime.h>

#define TT 512
#define BB 64
#define DD 512
#define HH 512
#define G4 2048   // 4*H

typedef __bf16 bf16_t;
typedef __attribute__((ext_vector_type(16))) __bf16 v16bf;
typedef __attribute__((ext_vector_type(4)))  __bf16 v4bf;
typedef __attribute__((ext_vector_type(8)))  float  v8f;

union FragBF { v16bf v; uint4 q[2]; };

// ---- fragment loaders ---------------------------------------------------
// A-matrix 16x32 bf16 as loaded here (contiguous k, not ISA table order): lane holds row m=lane%16; the 16
// bf16 per lane are 16 consecutive K values (matching B):
//   VGPR0-3 : K = k0 + 16*half .. +7     VGPR4-7 : K = k0 + 16*half + 8 .. +7
__device__ __forceinline__ v16bf load_fragA(const bf16_t* __restrict__ rowk0, int half) {
    FragBF f;
    f.q[0] = *reinterpret_cast<const uint4*>(rowk0 + 16 * half);
    f.q[1] = *reinterpret_cast<const uint4*>(rowk0 + 16 * half + 8);
    return f.v;
}
// B-matrix 32x16 bf16: lane holds column n=lane%16 with 16 consecutive K
// values starting at k0 + 16*half (one contiguous 32B stretch = two B128s).
// Column n of B = W.T column = row n of W (row-major) -> contiguous.
__device__ __forceinline__ v16bf load_fragB(const bf16_t* __restrict__ rowk) {
    FragBF f;
    f.q[0] = *reinterpret_cast<const uint4*>(rowk);
    f.q[1] = *reinterpret_cast<const uint4*>(rowk + 8);
    return f.v;
}

__device__ __forceinline__ float sigf(float x)     { return 1.0f / (1.0f + __expf(-x)); }
__device__ __forceinline__ float tanhfast(float x) { return 2.0f / (1.0f + __expf(-2.0f * x)) - 1.0f; }

// ---- big GEMM: C[M,N] = A[M,K](bf16) * W[N,K](bf16)^T, fp32 accum --------
// One wave computes a 64x64 block (4x4 WMMA tiles), K-loop in steps of 32.
__global__ __launch_bounds__(128) void gemm_bf16(const bf16_t* __restrict__ A,
                                                 const bf16_t* __restrict__ W,
                                                 float* __restrict__ C,
                                                 int M, int K, int N) {
    const int Nb   = N >> 6;
    const int gwid = blockIdx.x * (blockDim.x >> 5) + (threadIdx.x >> 5);
    if (gwid >= (M >> 6) * Nb) return;
    const int mblk = gwid / Nb, nblk = gwid % Nb;
    const int lane = threadIdx.x & 31, half = lane >> 4, lm = lane & 15;
    const int m0 = mblk << 6, n0 = nblk << 6;

    v8f acc[4][4] = {};
    const bf16_t* arow[4];
    const bf16_t* brow[4];
#pragma unroll
    for (int i = 0; i < 4; ++i) arow[i] = A + (size_t)(m0 + 16 * i + lm) * K;
#pragma unroll
    for (int j = 0; j < 4; ++j) brow[j] = W + (size_t)(n0 + 16 * j + lm) * K;

#pragma unroll 2
    for (int k0 = 0; k0 < K; k0 += 32) {
        v16bf af[4], bfr[4];
#pragma unroll
        for (int i = 0; i < 4; ++i) af[i] = load_fragA(arow[i] + k0, half);
#pragma unroll
        for (int j = 0; j < 4; ++j) bfr[j] = load_fragB(brow[j] + k0 + 16 * half);
#pragma unroll
        for (int i = 0; i < 4; ++i)
#pragma unroll
            for (int j = 0; j < 4; ++j)
                acc[i][j] = __builtin_amdgcn_wmma_f32_16x16x32_bf16(
                    false, af[i], false, bfr[j], (short)0, acc[i][j], false, false);
    }
    // C layout: lane -> n = lane%16, m = r + 8*half
#pragma unroll
    for (int i = 0; i < 4; ++i)
#pragma unroll
        for (int j = 0; j < 4; ++j) {
            float* crow = C + (size_t)(m0 + 16 * i + 8 * half) * N + (n0 + 16 * j + lm);
#pragma unroll
            for (int r = 0; r < 8; ++r) crow[(size_t)r * N] = acc[i][j][r];
        }
}

// ---- one recurrence step, both directions fused (blocks split by dir) ----
// wave -> (dir, 16-row batch tile, 16-col hidden tile); computes all 4 gate
// tiles so the LSTM cell fuses on the accumulators.
__global__ __launch_bounds__(128) void lstm_step(
    const bf16_t* __restrict__ hprev_fw, bf16_t* __restrict__ hnext_fw, float* __restrict__ cst_fw,
    const bf16_t* __restrict__ hprev_bw, bf16_t* __restrict__ hnext_bw, float* __restrict__ cst_bw,
    const bf16_t* __restrict__ whh_fw,   const bf16_t* __restrict__ whh_bw,
    const float* __restrict__ gx_fw,     const float* __restrict__ gx_bw,
    float* __restrict__ out_fw,          float* __restrict__ out_bw, int s) {
    const int wid  = blockIdx.x * (blockDim.x >> 5) + (threadIdx.x >> 5);  // 0..255
    const int dir  = wid >> 7;
    const int rem  = wid & 127;
    const int mt   = rem & 3, nt = rem >> 2;
    const int lane = threadIdx.x & 31, half = lane >> 4, lm = lane & 15;
    const int m0 = mt << 4, n0 = nt << 4;
    const int t  = dir ? (TT - 1 - s) : s;

    const bf16_t* hprev = dir ? hprev_bw : hprev_fw;
    bf16_t*       hnext = dir ? hnext_bw : hnext_fw;
    float*        cst   = dir ? cst_bw   : cst_fw;
    const bf16_t* whh   = dir ? whh_bw   : whh_fw;
    const float*  gx    = dir ? gx_bw    : gx_fw;
    float*        outp  = dir ? out_bw   : out_fw;

    v8f acc[4] = {};
    const bf16_t* arow = hprev + (size_t)(m0 + lm) * HH;
    const bf16_t* brow[4];
#pragma unroll
    for (int g = 0; g < 4; ++g) brow[g] = whh + (size_t)(g * HH + n0 + lm) * HH;

#pragma unroll 2
    for (int k0 = 0; k0 < HH; k0 += 32) {
        v16bf a = load_fragA(arow + k0, half);
#pragma unroll
        for (int g = 0; g < 4; ++g) {
            v16bf b = load_fragB(brow[g] + k0 + 16 * half);
            acc[g] = __builtin_amdgcn_wmma_f32_16x16x32_bf16(
                false, a, false, b, (short)0, acc[g], false, false);
        }
    }
    // fuse LSTM cell on accumulators: gates i,f,g,o at cols n, n+512, n+1024, n+1536
#pragma unroll
    for (int r = 0; r < 8; ++r) {
        const int m  = m0 + 8 * half + r;
        const int nl = n0 + lm;
        const size_t gxb = ((size_t)m * TT + t) * G4 + nl;   // Gx row = b*T + t
        const float vi = acc[0][r] + gx[gxb];
        const float vf = acc[1][r] + gx[gxb + 512];
        const float vg = acc[2][r] + gx[gxb + 1024];
        const float vo = acc[3][r] + gx[gxb + 1536];
        const int hn = m * HH + nl;
        const float c = sigf(vf) * cst[hn] + sigf(vi) * tanhfast(vg);
        cst[hn] = c;
        const float h = sigf(vo) * tanhfast(c);
        outp[(size_t)t * (BB * HH) + hn] = h;
        hnext[hn] = (bf16_t)h;
    }
}

// ---- helpers -------------------------------------------------------------
__global__ void cvt_f32_bf16(const float* __restrict__ src, bf16_t* __restrict__ dst, int n) {
    int i = (blockIdx.x * blockDim.x + threadIdx.x) * 4;
    if (i + 3 < n) {
        float4 v = *reinterpret_cast<const float4*>(src + i);
        v4bf o;
        o[0] = (bf16_t)v.x; o[1] = (bf16_t)v.y; o[2] = (bf16_t)v.z; o[3] = (bf16_t)v.w;
        *reinterpret_cast<v4bf*>(dst + i) = o;
    }
}

__global__ void init_state(bf16_t* hA_fw, bf16_t* hB_fw, bf16_t* hA_bw, bf16_t* hB_bw,
                           float* c_fw, float* c_bw) {
    int i = blockIdx.x * blockDim.x + threadIdx.x;  // 0..BB*HH-1
    bf16_t z = (bf16_t)0.0f;
    hA_fw[i] = z; hB_fw[i] = z; hA_bw[i] = z; hB_bw[i] = z;
    c_fw[i] = 0.0f; c_bw[i] = 0.0f;
}

__global__ void add_out(float* __restrict__ out, const float* __restrict__ add, int n) {
    int i = (blockIdx.x * blockDim.x + threadIdx.x) * 4;
    if (i + 3 < n) {
        float4 a = *reinterpret_cast<const float4*>(add + i);
        float4 o = *reinterpret_cast<float4*>(out + i);
        o.x += a.x; o.y += a.y; o.z += a.z; o.w += a.w;
        *reinterpret_cast<float4*>(out + i) = o;
    }
}

// ---- host ----------------------------------------------------------------
extern "C" void kernel_launch(void* const* d_in, const int* in_sizes, int n_in,
                              void* d_out, int out_size, void* d_ws, size_t ws_size,
                              hipStream_t stream) {
    const float* x      = (const float*)d_in[0];
    const float* wih_fw = (const float*)d_in[1];
    const float* whh_fw = (const float*)d_in[2];
    const float* wih_bw = (const float*)d_in[3];
    const float* whh_bw = (const float*)d_in[4];
    float* out = (float*)d_out;

    char* ws = (char*)d_ws;
    size_t off = 0;
    auto alloc = [&](size_t bytes) -> void* {
        void* p = ws + off;
        off += (bytes + 255) & ~(size_t)255;
        return p;
    };
    const size_t NX = (size_t)BB * TT * DD;   // 16.7M
    const size_t NW = (size_t)G4 * DD;        // 1.05M per weight
    const size_t NS = (size_t)BB * HH;        // 32768 state elems
    const size_t NO = (size_t)TT * BB * HH;   // 16.7M outputs

    bf16_t* x_bf     = (bf16_t*)alloc(NX * 2);
    bf16_t* wih_fw_b = (bf16_t*)alloc(NW * 2);
    bf16_t* whh_fw_b = (bf16_t*)alloc(NW * 2);
    bf16_t* wih_bw_b = (bf16_t*)alloc(NW * 2);
    bf16_t* whh_bw_b = (bf16_t*)alloc(NW * 2);
    float*  gx_fw    = (float*)alloc((size_t)BB * TT * G4 * 4);
    float*  gx_bw    = (float*)alloc((size_t)BB * TT * G4 * 4);
    float*  hbw_hist = (float*)alloc(NO * 4);
    bf16_t* hA_fw    = (bf16_t*)alloc(NS * 2);
    bf16_t* hB_fw    = (bf16_t*)alloc(NS * 2);
    bf16_t* hA_bw    = (bf16_t*)alloc(NS * 2);
    bf16_t* hB_bw    = (bf16_t*)alloc(NS * 2);
    float*  c_fw     = (float*)alloc(NS * 4);
    float*  c_bw     = (float*)alloc(NS * 4);
    (void)ws_size; (void)in_sizes; (void)n_in; (void)out_size;

    // 1) fp32 -> bf16 converts (no transposes needed by construction)
    cvt_f32_bf16<<<(int)(NX / 4 / 256), 256, 0, stream>>>(x, x_bf, (int)NX);
    cvt_f32_bf16<<<(int)(NW / 4 / 256), 256, 0, stream>>>(wih_fw, wih_fw_b, (int)NW);
    cvt_f32_bf16<<<(int)(NW / 4 / 256), 256, 0, stream>>>(whh_fw, whh_fw_b, (int)NW);
    cvt_f32_bf16<<<(int)(NW / 4 / 256), 256, 0, stream>>>(wih_bw, wih_bw_b, (int)NW);
    cvt_f32_bf16<<<(int)(NW / 4 / 256), 256, 0, stream>>>(whh_bw, whh_bw_b, (int)NW);
    init_state<<<(int)(NS / 256), 256, 0, stream>>>(hA_fw, hB_fw, hA_bw, hB_bw, c_fw, c_bw);

    // 2) hoisted input projections: Gx = x @ Wih.T  (WMMA bf16 GEMM)
    //    M=32768, K=512, N=2048 -> 512*32 waves of 64x64 blocks, 4 waves/block
    gemm_bf16<<<4096, 128, 0, stream>>>(x_bf, wih_fw_b, gx_fw, BB * TT, DD, G4);
    gemm_bf16<<<4096, 128, 0, stream>>>(x_bf, wih_bw_b, gx_bw, BB * TT, DD, G4);

    // 3) recurrence: 512 fused fw+bw steps (256 waves each)
    for (int s = 0; s < TT; ++s) {
        const bool even = (s & 1) == 0;
        lstm_step<<<64, 128, 0, stream>>>(
            even ? hA_fw : hB_fw, even ? hB_fw : hA_fw, c_fw,
            even ? hA_bw : hB_bw, even ? hB_bw : hA_bw, c_bw,
            whh_fw_b, whh_bw_b, gx_fw, gx_bw, out, hbw_hist, s);
    }

    // 4) out = h_fw + h_bw
    add_out<<<(int)(NO / 4 / 256), 256, 0, stream>>>(out, hbw_hist, (int)NO);
}
